// LambdaNetwork2D_27075473834286
// MI455X (gfx1250) — compile-verified, matrix operand order checked by reading, not run on hardware
//
#include <hip/hip_runtime.h>
#include <hip/hip_bf16.h>
#include <math.h>

typedef __attribute__((ext_vector_type(16))) __bf16 v16bf;
typedef __attribute__((ext_vector_type(8)))  float  v8f;
typedef __attribute__((ext_vector_type(4)))  float  f4;

#define NCOL 144          // column order: 0-63 q | 64-127 v | 128-143 k
#define EPSI 1e-3f
#define AST  264          // proj LDS stride (bf16 elems), 16B aligned rows, bank-spread
#define EST  1048         // E LDS stride (bf16 elems), 16B aligned rows, bank-spread

__device__ __forceinline__ unsigned short f2bf(float f) {
  union { float f; unsigned u; } x; x.f = f;
  unsigned r = x.u + 0x7FFFu + ((x.u >> 16) & 1u);   // round-to-nearest-even
  return (unsigned short)(r >> 16);
}

// ---------------------------------------------------------------- weights pack
__global__ void prep_w(const float* __restrict__ Wq, const float* __restrict__ Wk,
                       const float* __restrict__ Wv, unsigned short* __restrict__ Wt) {
  int col = blockIdx.x;     // 0..143
  int kc  = threadIdx.x;    // 0..255
  float w;
  if (col < 64)       w = Wq[kc * 64 + col];
  else if (col < 128) w = Wv[kc * 64 + (col - 64)];
  else                w = Wk[kc * 16 + (col - 128)];
  Wt[col * 256 + kc] = f2bf(w);   // [col][K] : K-contiguous B-fragment layout
}

// ------------------------------------------------------- qkv projection (WMMA)
__global__ __launch_bounds__(32) void proj_gemm(const float* __restrict__ x,
                                                const unsigned short* __restrict__ Wt,
                                                float* __restrict__ raw) {
  __shared__ unsigned short lA[16 * AST];
  int mt   = blockIdx.x;          // row tile (16 rows)
  int ct   = blockIdx.y;          // col tile (16 cols), 0..8
  int lane = threadIdx.x;
  // stage 16 rows x 256 channels of x as bf16
  const float* xb = x + (size_t)mt * 16 * 256;
  for (int i = 0; i < 32; ++i) {
    int f = lane + i * 32;               // float4 index 0..1023
    f4 d  = *(const f4*)(xb + (size_t)f * 4);
    int r = (f * 4) >> 8, c = (f * 4) & 255;
    unsigned short* p = &lA[r * AST + c];
    p[0] = f2bf(d.x); p[1] = f2bf(d.y); p[2] = f2bf(d.z); p[3] = f2bf(d.w);
  }
  __syncthreads();
  int kh = lane >> 4, l16 = lane & 15;
  const unsigned short* wb = Wt + (ct * 16 + l16) * 256 + kh * 16;
  const unsigned short* ab = &lA[l16 * AST + kh * 16];
  v8f acc = (v8f){0.f,0.f,0.f,0.f,0.f,0.f,0.f,0.f};
#pragma unroll
  for (int k0 = 0; k0 < 256; k0 += 32) {
    v16bf a = *(const v16bf*)(ab + k0);
    v16bf b = *(const v16bf*)(wb + k0);
    acc = __builtin_amdgcn_wmma_f32_16x16x32_bf16(false, a, false, b,
                                                  (short)0, acc, false, false);
  }
#pragma unroll
  for (int r = 0; r < 8; ++r) {
    int row = mt * 16 + r + 8 * kh;                 // D layout: vgpr r -> M = r + 8*(lane>=16)
    raw[(size_t)row * NCOL + ct * 16 + l16] = acc[r];
  }
}

// ----------------------------------------------------------------- BN on q / v
__global__ void bn_qv(const float* __restrict__ raw,
                      const float* gq, const float* bq, const float* mq, const float* vq,
                      const float* gv, const float* bv, const float* mv, const float* vvar,
                      float* __restrict__ q_bn, float* __restrict__ v_bn,
                      unsigned short* __restrict__ Vt) {
  int idx = blockIdx.x * 256 + threadIdx.x;    // over 16384*128
  int row = idx >> 7;
  int c   = idx & 127;
  float t = raw[(size_t)row * NCOL + c];
  if (c < 64) {
    float o = gq[c] * (t - mq[c]) * rsqrtf(vq[c] + EPSI) + bq[c];
    q_bn[row * 64 + c] = o;
  } else {
    int d   = c - 64;
    float o = gv[d] * (t - mv[d]) * rsqrtf(vvar[d] + EPSI) + bv[d];
    v_bn[row * 64 + d] = o;
    int b = row >> 10, m = row & 1023;
    Vt[((b * 64 + d) << 10) + m] = f2bf(o);    // [col=(b,v)][m] K-contiguous
  }
}

// -------------------------------------------------------- softmax over k chans
__global__ void softmax_k(const float* __restrict__ raw, float* __restrict__ ks) {
  int row = blockIdx.x * 256 + threadIdx.x;    // 16384 rows total
  const float* p = raw + (size_t)row * NCOL + 128;
  float v[16], mx = -3.4e38f;
#pragma unroll
  for (int i = 0; i < 16; ++i) { v[i] = p[i]; mx = fmaxf(mx, v[i]); }
  float s = 0.f;
#pragma unroll
  for (int i = 0; i < 16; ++i) { v[i] = __expf(v[i] - mx); s += v[i]; }
  float inv = 1.f / s;
#pragma unroll
  for (int i = 0; i < 16; ++i) ks[row * 16 + i] = v[i] * inv;
}

// --------------------------------------------------- lam_c[b,k,v] = k^T . v
__global__ void lamc_k(const float* __restrict__ ks, const float* __restrict__ v_bn,
                       float* __restrict__ lamc) {
  __shared__ float sk[64 * 16];
  __shared__ float sv[64 * 64];
  int b = blockIdx.x, t = threadIdx.x;
  float acc[4] = {0.f, 0.f, 0.f, 0.f};
  for (int n0 = 0; n0 < 1024; n0 += 64) {
    __syncthreads();
    const float* kg = ks   + (size_t)((b << 10) + n0) * 16;
    const float* vg = v_bn + (size_t)((b << 10) + n0) * 64;
#pragma unroll
    for (int i = 0; i < 4; ++i)  sk[t + i * 256] = kg[t + i * 256];
#pragma unroll
    for (int i = 0; i < 16; ++i) sv[t + i * 256] = vg[t + i * 256];
    __syncthreads();
#pragma unroll
    for (int j = 0; j < 4; ++j) {
      int p = t + j * 256, kk = p >> 6, vv = p & 63;
      float a = 0.f;
      for (int n = 0; n < 64; ++n) a += sk[n * 16 + kk] * sv[n * 64 + vv];
      acc[j] += a;
    }
  }
#pragma unroll
  for (int j = 0; j < 4; ++j) lamc[(b << 10) + t + j * 256] = acc[j];
}

// -------- main: lam_p GEMM (WMMA) fused with +lam_c and q-contraction + output
__global__ __launch_bounds__(128, 1) void lambda_main(const float* __restrict__ E,
                                                      const unsigned short* __restrict__ Vt,
                                                      const float* __restrict__ q_bn,
                                                      const float* __restrict__ lamc,
                                                      float* __restrict__ out) {
  extern __shared__ unsigned short lE[];       // 4 n's x [16 k][EST m] bf16
  int nt  = blockIdx.x;                        // n tile of 4
  int tid = threadIdx.x;
  // stage E[n, m, k] -> lE[nn][k][m] bf16 (transpose + convert, read-once fp32)
  for (int nn = 0; nn < 4; ++nn) {
    const float* eb = E + (size_t)(nt * 4 + nn) * (1024 * 16);
    for (int i = 0; i < 32; ++i) {
      int f = tid + i * 128;                   // float4 index 0..4095
      f4 d  = *(const f4*)(eb + (size_t)f * 4);
      int m = f >> 2, kb = (f & 3) * 4;
      lE[(nn * 16 + kb + 0) * EST + m] = f2bf(d.x);
      lE[(nn * 16 + kb + 1) * EST + m] = f2bf(d.y);
      lE[(nn * 16 + kb + 2) * EST + m] = f2bf(d.z);
      lE[(nn * 16 + kb + 3) * EST + m] = f2bf(d.w);
    }
  }
  __syncthreads();

  int wave = tid >> 5, lane = tid & 31, kh = lane >> 4, l16 = lane & 15;
  for (int bi = 0; bi < 4; ++bi) {
    int b = wave * 4 + bi;
    v8f acc[4][4];
#pragma unroll
    for (int i = 0; i < 4; ++i)
#pragma unroll
      for (int j = 0; j < 4; ++j)
        acc[i][j] = (v8f){0.f,0.f,0.f,0.f,0.f,0.f,0.f,0.f};

    const unsigned short* vb = Vt + (size_t)(b * 64) * 1024;
    for (int m0 = 0; m0 < 1024; m0 += 32) {
      v16bf bf[4];
#pragma unroll
      for (int vt = 0; vt < 4; ++vt)
        bf[vt] = *(const v16bf*)(vb + (size_t)((vt * 16 + l16) * 1024) + m0 + kh * 16);
#pragma unroll
      for (int nn = 0; nn < 4; ++nn) {
        v16bf af = *(const v16bf*)(&lE[(nn * 16 + l16) * EST + m0 + kh * 16]);
#pragma unroll
        for (int vt = 0; vt < 4; ++vt)
          acc[nn][vt] = __builtin_amdgcn_wmma_f32_16x16x32_bf16(false, af, false, bf[vt],
                                                                (short)0, acc[nn][vt],
                                                                false, false);
      }
    }
    // epilogue: (lam_p + lam_c) contracted with q over k; write Y
#pragma unroll
    for (int vt = 0; vt < 4; ++vt) {
      float lc[8];
#pragma unroll
      for (int r = 0; r < 8; ++r)
        lc[r] = lamc[(b << 10) + (r + 8 * kh) * 64 + vt * 16 + l16];
#pragma unroll
      for (int nn = 0; nn < 4; ++nn) {
        int n = nt * 4 + nn;
        const float* qr = q_bn + (size_t)((b << 10) + n) * 64;
#pragma unroll
        for (int h = 0; h < 4; ++h) {
          float part = 0.f;
#pragma unroll
          for (int r = 0; r < 8; ++r)
            part += qr[h * 16 + r + 8 * kh] * (acc[nn][vt][r] + lc[r]);
          part += __shfl_xor(part, 16, 32);    // merge complementary k-half
          if (kh == 0)
            out[((size_t)(b << 10) + n) * 256 + h * 64 + vt * 16 + l16] = part;
        }
      }
    }
  }
}

// ------------------------------------------------------------------- launcher
extern "C" void kernel_launch(void* const* d_in, const int* in_sizes, int n_in,
                              void* d_out, int out_size, void* d_ws, size_t ws_size,
                              hipStream_t stream) {
  const float* x  = (const float*)d_in[0];
  const float* Wq = (const float*)d_in[1];
  const float* Wk = (const float*)d_in[2];
  const float* Wv = (const float*)d_in[3];
  const float* gq = (const float*)d_in[4];
  const float* bq = (const float*)d_in[5];
  const float* mq = (const float*)d_in[6];
  const float* vq = (const float*)d_in[7];
  const float* gv = (const float*)d_in[8];
  const float* bv = (const float*)d_in[9];
  const float* mv = (const float*)d_in[10];
  const float* vv = (const float*)d_in[11];
  const float* E  = (const float*)d_in[12];

  char* ws = (char*)d_ws;
  unsigned short* Wt  = (unsigned short*)(ws + 0);         //  73,728 B
  float*          raw = (float*)(ws + 73728);              // 9,437,184 B
  float*          qbn = (float*)(ws + 9510912);            // 4,194,304 B
  float*          vbn = (float*)(ws + 13705216);           // 4,194,304 B
  float*          ksm = (float*)(ws + 17899520);           // 1,048,576 B
  unsigned short* Vt  = (unsigned short*)(ws + 18948096);  // 2,097,152 B
  float*          lmc = (float*)(ws + 21045248);           //    65,536 B
  float* out = (float*)d_out;

  prep_w     <<<144, 256, 0, stream>>>(Wq, Wk, Wv, Wt);
  proj_gemm  <<<dim3(1024, 9), 32, 0, stream>>>(x, Wt, raw);
  bn_qv      <<<8192, 256, 0, stream>>>(raw, gq, bq, mq, vq, gv, bv, mv, vv, qbn, vbn, Vt);
  softmax_k  <<<64, 256, 0, stream>>>(raw, ksm);
  lamc_k     <<<16, 256, 0, stream>>>(ksm, vbn, lmc);
  lambda_main<<<256, 128, (size_t)(4 * 16 * EST * sizeof(unsigned short)), stream>>>(E, Vt, qbn, lmc, out);
}
